// Spec2Roll_67843303408361
// MI455X (gfx1250) — matrix operand order repaired and checked by reading, not compiled, not run on hardware
//
#include <hip/hip_runtime.h>
#include <hip/hip_bf16.h>

// ---------------------------------------------------------------------------
// Problem constants: B=2, S=2048, IN=512, OUT=768, K=31, GROUPS=6 -> d=128,
// pad=15, padded seq KV=2078 (stored as 2080 rows per batch).
// ---------------------------------------------------------------------------
constexpr int Bn   = 2;
constexpr int Sn   = 2048;
constexpr int INn  = 512;
constexpr int OUTn = 768;
constexpr int Kn   = 31;
constexpr int GR   = 6;
constexpr int Dg   = OUTn / GR;          // 128
constexpr int PADn = (Kn - 1) / 2;       // 15
constexpr int KV   = Sn + 2 * PADn;      // 2078
constexpr int KVP  = 2080;               // padded rows per batch (32-multiple)

// ---------------------------------------------------------------------------
// Workspace layout.  f32 region (element offsets), then bf16 regions (bytes).
//   q     f32 [4096 x 768]
//   k_pad f32 [2*2080 x 768]
//   v_pad f32 [2*2080 x 768]
//   xpad  bf16 hi/lo [2*2080 x 512]   (zero rows realize jnp.pad)
//   Wq/Wk/Wv bf16 hi/lo [768 x 512]   total ~51.4 MB
// ---------------------------------------------------------------------------
constexpr size_t QOFF_F   = 0;
constexpr size_t KOFF_F   = (size_t)Bn * Sn * OUTn;              // 3145728
constexpr size_t VOFF_F   = KOFF_F + (size_t)Bn * KVP * OUTn;    // 6340608
constexpr size_t F32_B    = (VOFF_F + (size_t)Bn * KVP * OUTn) * 4; // 38141952
constexpr size_t XPROWS   = (size_t)Bn * KVP;                    // 4160
constexpr size_t XP_B     = XPROWS * INn * 2;                    // 4259840
constexpr size_t W_B      = (size_t)OUTn * INn * 2;              // 786432
constexpr size_t XPH_OFF  = F32_B;
constexpr size_t XPL_OFF  = XPH_OFF + XP_B;
constexpr size_t WQH_OFF  = XPL_OFF + XP_B;
constexpr size_t WQL_OFF  = WQH_OFF + W_B;
constexpr size_t WKH_OFF  = WQL_OFF + W_B;
constexpr size_t WKL_OFF  = WKH_OFF + W_B;
constexpr size_t WVH_OFF  = WKL_OFF + W_B;
constexpr size_t WVL_OFF  = WVH_OFF + W_B;

typedef __attribute__((ext_vector_type(4)))  float  f4;
typedef __attribute__((ext_vector_type(8)))  float  v8f;
typedef __attribute__((ext_vector_type(8)))  __bf16 v8bf;
typedef __attribute__((ext_vector_type(16))) __bf16 v16bf;

__device__ __forceinline__ f4 ldf4(const float* p) {
    return *reinterpret_cast<const f4*>(p);
}
__device__ __forceinline__ v8bf ld8bf(const __bf16* p) {
    return *reinterpret_cast<const v8bf*>(p);
}
// A fragment: lane half h owns K = kk+16h..+15
__device__ __forceinline__ v16bf ldAfrag(const __bf16* row, int kk, int h) {
    v8bf x0 = ld8bf(row + kk + 16 * h);
    v8bf x1 = ld8bf(row + kk + 16 * h + 8);
    return __builtin_shufflevector(x0, x1, 0,1,2,3,4,5,6,7,8,9,10,11,12,13,14,15);
}
// B fragment: 16 consecutive K values
__device__ __forceinline__ v16bf ldBfrag(const __bf16* p) {
    v8bf x0 = ld8bf(p);
    v8bf x1 = ld8bf(p + 8);
    return __builtin_shufflevector(x0, x1, 0,1,2,3,4,5,6,7,8,9,10,11,12,13,14,15);
}

// ---------------------------------------------------------------------------
// Prep A: split+pad x -> xpad_hi/lo (bf16, zero rows outside valid range).
// One thread per 8 elements.
// ---------------------------------------------------------------------------
__global__ __launch_bounds__(256)
void split_pad_x(const float* __restrict__ x,
                 __bf16* __restrict__ hi, __bf16* __restrict__ lo) {
    const int idx = blockIdx.x * 256 + threadIdx.x;     // < 4160*512/8
    const int e0  = idx * 8;
    const int r   = e0 / INn;
    const int c   = e0 - r * INn;
    const int b   = r / KVP;
    const int t   = r - b * KVP;
    const int xr  = t - PADn;
    const bool valid = (xr >= 0) && (xr < Sn) && (t < KV);

    v8bf h8 = (v8bf)(__bf16)0.0f, l8 = h8;
    if (valid) {
        const float* src = x + ((size_t)(b * Sn + xr)) * INn + c;
        const f4 f0 = ldf4(src), f1 = ldf4(src + 4);
#pragma unroll
        for (int i = 0; i < 4; ++i) {
            __bf16 hh = (__bf16)f0[i];
            h8[i] = hh; l8[i] = (__bf16)(f0[i] - (float)hh);
            __bf16 hh2 = (__bf16)f1[i];
            h8[4 + i] = hh2; l8[4 + i] = (__bf16)(f1[i] - (float)hh2);
        }
    }
    *reinterpret_cast<v8bf*>(hi + e0) = h8;
    *reinterpret_cast<v8bf*>(lo + e0) = l8;
}

// Prep B: split a weight matrix (768x512) -> bf16 hi/lo. Thread per 8 elems.
__global__ __launch_bounds__(256)
void split_w(const float* __restrict__ w,
             __bf16* __restrict__ hi, __bf16* __restrict__ lo) {
    const int e0 = (blockIdx.x * 256 + threadIdx.x) * 8;  // < 768*512
    const f4 f0 = ldf4(w + e0), f1 = ldf4(w + e0 + 4);
    v8bf h8, l8;
#pragma unroll
    for (int i = 0; i < 4; ++i) {
        __bf16 hh = (__bf16)f0[i];
        h8[i] = hh; l8[i] = (__bf16)(f0[i] - (float)hh);
        __bf16 hh2 = (__bf16)f1[i];
        h8[4 + i] = hh2; l8[4 + i] = (__bf16)(f1[i] - (float)hh2);
    }
    *reinterpret_cast<v8bf*>(hi + e0) = h8;
    *reinterpret_cast<v8bf*>(lo + e0) = l8;
}

// ---------------------------------------------------------------------------
// QKV GEMM, pure-WMMA hot loop. One wave -> 32(M) x 64(N) output tile.
// 2-term bf16 split: acc += Ahi*Bhi + Ahi*Blo + Alo*Bhi  (f32-class accuracy).
// Per 32-K step: 24 b128 loads, 24 v_wmma, ~no VALU.
//   blockIdx.x : N block (64 cols) [0,12)
//   blockIdx.y : M tile (32 rows over padded rows) [0,130)  (q uses [0,128))
//   blockIdx.z : 0=q, 1=k, 2=v
// ---------------------------------------------------------------------------
__global__ __launch_bounds__(32)
void qkv_gemm_wmma(const __bf16* __restrict__ xph, const __bf16* __restrict__ xpl,
                   const __bf16* __restrict__ wqh, const __bf16* __restrict__ wql,
                   const __bf16* __restrict__ wkh, const __bf16* __restrict__ wkl,
                   const __bf16* __restrict__ wvh, const __bf16* __restrict__ wvl,
                   float* __restrict__ q, float* __restrict__ kp,
                   float* __restrict__ vp) {
    const int lane = threadIdx.x & 31;
    const int h    = lane >> 4;
    const int m    = lane & 15;
    const int nb   = blockIdx.x;
    const int ty   = blockIdx.y;
    const int mat  = blockIdx.z;

    if (mat == 0 && ty >= (Bn * Sn) / 32) return;

    const __bf16* wh = (mat == 0) ? wqh : (mat == 1 ? wkh : wvh);
    const __bf16* wl = (mat == 0) ? wql : (mat == 1 ? wkl : wvl);

    // Per-lane A row (padded-row index). q rows sit at offset PAD inside xpad.
    size_t prow0, prow1;  // m-tile 0 (rows +0..15) and m-tile 1 (rows +16..31)
    {
        const int r0 = ty * 32 + m;       // logical output row (tile 0)
        const int r1 = r0 + 16;           // tile 1
        if (mat == 0) {
            prow0 = (size_t)(r0 >> 11) * KVP + PADn + (r0 & (Sn - 1));
            prow1 = (size_t)(r1 >> 11) * KVP + PADn + (r1 & (Sn - 1));
        } else {
            prow0 = (size_t)r0;
            prow1 = (size_t)r1;
        }
    }
    const __bf16* a0h = xph + prow0 * INn;
    const __bf16* a0l = xpl + prow0 * INn;
    const __bf16* a1h = xph + prow1 * INn;
    const __bf16* a1l = xpl + prow1 * INn;

    const int n0 = nb * 64;

    v8f acc[2][4];
#pragma unroll
    for (int mt = 0; mt < 2; ++mt)
#pragma unroll
        for (int ti = 0; ti < 4; ++ti) acc[mt][ti] = v8f{0};

    for (int kk = 0; kk < INn; kk += 32) {
        __builtin_prefetch(a0h + kk + 32, 0, 0);  // global_prefetch_b8
        const v16bf ahi0 = ldAfrag(a0h, kk, h);
        const v16bf alo0 = ldAfrag(a0l, kk, h);
        const v16bf ahi1 = ldAfrag(a1h, kk, h);
        const v16bf alo1 = ldAfrag(a1l, kk, h);

#pragma unroll
        for (int ti = 0; ti < 4; ++ti) {
            const size_t wro = (size_t)(n0 + 16 * ti + m) * INn + kk + 16 * h;
            const v16bf bhi = ldBfrag(wh + wro);
            const v16bf blo = ldBfrag(wl + wro);

            acc[0][ti] = __builtin_amdgcn_wmma_f32_16x16x32_bf16(
                false, ahi0, false, bhi, (short)0, acc[0][ti], false, false);
            acc[0][ti] = __builtin_amdgcn_wmma_f32_16x16x32_bf16(
                false, ahi0, false, blo, (short)0, acc[0][ti], false, false);
            acc[0][ti] = __builtin_amdgcn_wmma_f32_16x16x32_bf16(
                false, alo0, false, bhi, (short)0, acc[0][ti], false, false);

            acc[1][ti] = __builtin_amdgcn_wmma_f32_16x16x32_bf16(
                false, ahi1, false, bhi, (short)0, acc[1][ti], false, false);
            acc[1][ti] = __builtin_amdgcn_wmma_f32_16x16x32_bf16(
                false, ahi1, false, blo, (short)0, acc[1][ti], false, false);
            acc[1][ti] = __builtin_amdgcn_wmma_f32_16x16x32_bf16(
                false, alo1, false, bhi, (short)0, acc[1][ti], false, false);
        }
    }

    // C/D layout: VGPR rr -> M = rr + 8*h, N = lane%16
    float* dst = (mat == 0) ? q : (mat == 1 ? kp : vp);
#pragma unroll
    for (int mt = 0; mt < 2; ++mt) {
#pragma unroll
        for (int ti = 0; ti < 4; ++ti) {
#pragma unroll
            for (int rr = 0; rr < 8; ++rr) {
                const int grow = ty * 32 + mt * 16 + rr + 8 * h;
                const int gcol = n0 + 16 * ti + m;
                dst[(size_t)grow * OUTn + gcol] = acc[mt][ti][rr];
            }
        }
    }
}

// ---------------------------------------------------------------------------
// Banded windowed attention. One wave per (b,s,g); lane owns 4 of 128 dims.
// ---------------------------------------------------------------------------
__global__ __launch_bounds__(256)
void window_attn(const float* __restrict__ ws,
                 const float* __restrict__ rel,
                 float* __restrict__ out) {
    const int lane = threadIdx.x & 31;
    const int wid  = threadIdx.x >> 5;
    const int item = blockIdx.x * 8 + wid;  // < 24576
    const int g = item % GR;
    const int s = (item / GR) % Sn;
    const int b = item / (GR * Sn);

    const float* q  = ws + QOFF_F;
    const float* kp = ws + KOFF_F;
    const float* vp = ws + VOFF_F;

    const int o0 = g * Dg + 4 * lane;
    const f4 q4 = ldf4(q + (size_t)(b * Sn + s) * OUTn + o0);

    float e[Kn];
    const float* krow = kp + (size_t)(b * KVP + s) * OUTn + o0;
#pragma unroll
    for (int k = 0; k < Kn; ++k) {
        const f4 kv = ldf4(krow + (size_t)k * OUTn);
        e[k] = q4[0] * kv[0] + q4[1] * kv[1] + q4[2] * kv[2] + q4[3] * kv[3];
    }
#pragma unroll
    for (int j = 0; j < 4; ++j) {
        const float* rr = rel + (size_t)(o0 + j) * Kn;
        const float qj = q4[j];
#pragma unroll
        for (int k = 0; k < Kn; ++k) e[k] += qj * rr[k];
    }

#pragma unroll
    for (int off = 16; off > 0; off >>= 1) {
#pragma unroll
        for (int k = 0; k < Kn; ++k) e[k] += __shfl_xor(e[k], off, 32);
    }

    float mx = e[0];
#pragma unroll
    for (int k = 1; k < Kn; ++k) mx = fmaxf(mx, e[k]);
    float a[Kn], sum = 0.0f;
#pragma unroll
    for (int k = 0; k < Kn; ++k) { a[k] = __expf(e[k] - mx); sum += a[k]; }
    const float inv = 1.0f / sum;
#pragma unroll
    for (int k = 0; k < Kn; ++k) a[k] *= inv;

    f4 o4 = f4{0.0f, 0.0f, 0.0f, 0.0f};
    const float* vrow = vp + (size_t)(b * KVP + s) * OUTn + o0;
#pragma unroll
    for (int k = 0; k < Kn; ++k) {
        const f4 vv = ldf4(vrow + (size_t)k * OUTn);
        o4 += a[k] * vv;
    }
    *reinterpret_cast<f4*>(out + (size_t)(b * Sn + s) * OUTn + o0) = o4;

    float av = a[0];
#pragma unroll
    for (int k = 1; k < Kn; ++k) av = (lane == k) ? a[k] : av;
    if (lane < Kn) {
        out[(size_t)Bn * Sn * OUTn +
            ((size_t)(b * Sn + s) * GR + g) * Kn + lane] = av;
    }
}

extern "C" void kernel_launch(void* const* d_in, const int* in_sizes, int n_in,
                              void* d_out, int out_size, void* d_ws, size_t ws_size,
                              hipStream_t stream) {
    const float* x   = (const float*)d_in[0];
    const float* Wq  = (const float*)d_in[1];
    const float* Wk  = (const float*)d_in[2];
    const float* Wv  = (const float*)d_in[3];
    const float* rel = (const float*)d_in[4];
    float* out = (float*)d_out;
    char*  ws  = (char*)d_ws;

    float* qf  = (float*)(ws);               // uses QOFF_F/KOFF_F/VOFF_F
    float* kpf = (float*)(ws) + KOFF_F;
    float* vpf = (float*)(ws) + VOFF_F;
    __bf16* xph = (__bf16*)(ws + XPH_OFF);
    __bf16* xpl = (__bf16*)(ws + XPL_OFF);
    __bf16* wqh = (__bf16*)(ws + WQH_OFF);
    __bf16* wql = (__bf16*)(ws + WQL_OFF);
    __bf16* wkh = (__bf16*)(ws + WKH_OFF);
    __bf16* wkl = (__bf16*)(ws + WKL_OFF);
    __bf16* wvh = (__bf16*)(ws + WVH_OFF);
    __bf16* wvl = (__bf16*)(ws + WVL_OFF);

    // Prep: split/pad to bf16 hi/lo (one-time, streamed; ~13 MB written).
    split_pad_x<<<(int)(XPROWS * INn / 8 / 256), 256, 0, stream>>>(x, xph, xpl);
    const int wblk = OUTn * INn / 8 / 256;   // 192
    split_w<<<wblk, 256, 0, stream>>>(Wq, wqh, wql);
    split_w<<<wblk, 256, 0, stream>>>(Wk, wkh, wkl);
    split_w<<<wblk, 256, 0, stream>>>(Wv, wvh, wvl);

    // QKV projection: grid (12, 130, 3), one wave per block.
    dim3 g1(OUTn / 64, (Bn * KVP) / 32, 3);
    qkv_gemm_wmma<<<g1, 32, 0, stream>>>(xph, xpl, wqh, wql, wkh, wkl,
                                         wvh, wvl, qf, kpf, vpf);

    // Windowed attention: 24576 waves, 8 per block.
    window_attn<<<(Bn * Sn * GR) / 8, 256, 0, stream>>>((float*)ws, rel, out);
}
